// VTF_CNNs_pre_freeze_81063212745094
// MI455X (gfx1250) — matrix operand order repaired and checked by reading, not compiled, not run on hardware
//
#include <hip/hip_runtime.h>
#include <hip/hip_bf16.h>

// ---------------------------------------------------------------------------
// MI455X (gfx1250) ResNet18x2 + fuse.
//   - NHWC activations, (r,s,ci) K-order packed f16 weights
//   - convs = implicit GEMM on v_wmma_f32_16x16x32_f16
//   - block tile 16(M) x 256(N): 4 waves x (16x64 per wave, 4 accumulators)
//   - A (weights) double-buffered through LDS via global_load_async_to_lds
//   - B fragments are two aligned b128 loads (channels-innermost => no gather)
//   - kernel templated on (R,S); Cin power-of-two => shift/mask K decompose
//   - BN (eval) + residual + ReLU fused epilogue, vectorized b128 stores
// ---------------------------------------------------------------------------

typedef __attribute__((ext_vector_type(16))) _Float16 v16h;
typedef __attribute__((ext_vector_type(8)))  _Float16 v8h;
typedef __attribute__((ext_vector_type(8)))  float    v8f;
typedef __attribute__((ext_vector_type(4)))  int      v4i;

#define AS_GLOBAL __attribute__((address_space(1)))
#define AS_LDS    __attribute__((address_space(3)))

#if __has_builtin(__builtin_amdgcn_global_load_async_to_lds_b128)
#define ASYNC_A 1
// src: global 128-bit vector ptr, dst: LDS 128-bit vector ptr, offset, cpol
static __device__ __forceinline__ void async_copy_b128(const _Float16* src,
                                                       _Float16* dst) {
  AS_GLOBAL void* gs = (AS_GLOBAL void*)src;
  AS_LDS    void* ld = (AS_LDS void*)dst;
  __builtin_amdgcn_global_load_async_to_lds_b128((AS_GLOBAL v4i*)gs,
                                                 (AS_LDS v4i*)ld, 0, 0);
}
#else
#define ASYNC_A 0
#endif

// ---------------------------------------------------------------------------
// rgb [8,3,224,224] fp32 -> NHWC f16 [8,224,224,16] with channels 3..15 = 0
// ---------------------------------------------------------------------------
__global__ void cvt_input_kernel(const float* __restrict__ src,
                                 _Float16* __restrict__ dst) {
  const int total = 8 * 224 * 224;
  int i = blockIdx.x * blockDim.x + threadIdx.x;
  if (i >= total) return;
  int w = i % 224; int t = i / 224;
  int h = t % 224;
  int b = t / 224;
#pragma unroll
  for (int c = 0; c < 16; ++c) {
    float v = (c < 3) ? src[((b * 3 + c) * 224 + h) * 224 + w] : 0.f;
    dst[(long)i * 16 + c] = (_Float16)v;
  }
}

// ---------------------------------------------------------------------------
// pack weight [O, CinReal, R, S] fp32 -> [O, Kpad] f16 with K = (krs, ci),
// ci padded to CinPad, trailing K zero-padded.
// ---------------------------------------------------------------------------
__global__ void pack_w_kernel(const float* __restrict__ W,
                              _Float16* __restrict__ Wp,
                              int CinReal, int CinPad, int RS, int Kpad,
                              long total) {
  long i = (long)blockIdx.x * blockDim.x + threadIdx.x;
  if (i >= total) return;
  int o   = (int)(i / Kpad);
  int k   = (int)(i - (long)o * Kpad);
  int krs = k / CinPad;
  int ci  = k - krs * CinPad;
  float v = 0.f;
  if (krs < RS && ci < CinReal)
    v = W[((long)o * CinReal + ci) * RS + krs];
  Wp[i] = (_Float16)v;
}

// ---------------------------------------------------------------------------
// Implicit-GEMM conv, NHWC, templated on filter size.
//   grid.y = Cout/16 (mtile), grid.x covers ceil(N/16)/16 (16 n-tiles/block)
//   block = 128 threads = 4 waves; wave w owns n-subtiles 4w..4w+3.
// A layout (16-bit 16x32): lane->row m=lane&15, half=lane>>4;
//   elems i = K[16*half + i]
// B layout (16-bit 32x16): lane->col n=lane&15; elems i = K[16*half + i]
// C layout: vgpr r -> m = r + 8*half, n = lane&15
// ---------------------------------------------------------------------------
template <int R, int S>
__global__ __launch_bounds__(128) void conv_wmma_kernel(
    const _Float16* __restrict__ X,   // [8, IH, IW, Cin] f16
    const _Float16* __restrict__ Wp,  // [Cout, Kpad] f16 packed
    const float*    __restrict__ gamma,
    const float*    __restrict__ beta,
    const _Float16* __restrict__ Res, // [8, OH, OW, Cout] or nullptr
    _Float16*       __restrict__ Y,   // [8, OH, OW, Cout]
    int cinLog2, int IH, int IW, int Cout, int OH, int OW,
    int stride, int pad, int Kprod, int Kpad, int relu)
{
  const int Cin = 1 << cinLog2;
  __shared__ alignas(16) _Float16 lds_a[2][16 * 32];  // double-buffered 2KB

  const int mtile = blockIdx.y;
  const int Ntot = 8 * OH * OW;
  const int tid  = threadIdx.x;
  const int wave = tid >> 5;
  const int lane = tid & 31;
  const int half = lane >> 4;
  const int l15  = lane & 15;

  // ---- per-subtile output-pixel decomposition (4 subtiles per wave) ------
  int bimg[4], ohv[4], owv[4], ih0[4], iw0[4];
  bool nv[4];
  const int hw = OH * OW;
#pragma unroll
  for (int t = 0; t < 4; ++t) {
    int n = (blockIdx.x * 16 + wave * 4 + t) * 16 + l15;
    nv[t] = (n < Ntot);
    int nn = nv[t] ? n : 0;
    bimg[t] = nn / hw;
    int rem = nn - bimg[t] * hw;
    ohv[t] = rem / OW;
    owv[t] = rem - ohv[t] * OW;
    ih0[t] = ohv[t] * stride - pad;
    iw0[t] = owv[t] * stride - pad;
  }

  v8f acc[4] = {};
  const int arow = tid >> 2;           // staging: thread -> (row, 8-elem seg)
  const int aseg = tid & 3;
  const _Float16* Asrc = Wp + (long)(mtile * 16 + arow) * Kpad + aseg * 8;
  _Float16* Adst0 = &lds_a[0][arow * 32 + aseg * 8];
  _Float16* Adst1 = &lds_a[1][arow * 32 + aseg * 8];

  const int nchunks = Kpad >> 5;

  // ---- stage chunk 0 ------------------------------------------------------
  if (tid < 64) {
#if ASYNC_A
    async_copy_b128(Asrc, Adst0);
#else
    *(v8h*)Adst0 = *(const v8h*)Asrc;
#endif
  }

  for (int c = 0; c < nchunks; ++c) {
#if ASYNC_A
    asm volatile("s_wait_asynccnt 0" ::: "memory");
#endif
    __syncthreads();  // chunk c resident in lds_a[c&1]

    const _Float16* abuf = lds_a[c & 1];
    v8h alo = *(const v8h*)(abuf + l15 * 32 + 16 * half);
    v8h ahi = *(const v8h*)(abuf + l15 * 32 + 16 * half + 8);
    v16h a;
#pragma unroll
    for (int i = 0; i < 8; ++i) { a[i] = alo[i]; a[8 + i] = ahi[i]; }

    // stage chunk c+1 into the other buffer (overlaps with WMMA below)
    if (c + 1 < nchunks && tid < 64) {
      const _Float16* src = Asrc + ((c + 1) << 5);
      _Float16* dst = ((c + 1) & 1) ? Adst1 : Adst0;
#if ASYNC_A
      async_copy_b128(src, dst);
#else
      *(v8h*)dst = *(const v8h*)src;
      if (c + 2 < nchunks) __builtin_prefetch(Asrc + ((c + 2) << 5), 0, 1);
#endif
    }

    // ---- K decomposition: shift/mask (Cin pow2), const small-div by S ----
    const int kb  = (c << 5) + 16 * half;
    const int krs = kb >> cinLog2;
    const int ci  = kb & (Cin - 1);
    const int r   = krs / S;
    const int s   = krs - r * S;
    const bool kvalid = (kb < Kprod);

#pragma unroll
    for (int t = 0; t < 4; ++t) {
      v16h b = {};
      if (kvalid && nv[t]) {
        const int ih = ih0[t] + r, iw = iw0[t] + s;
        if ((unsigned)ih < (unsigned)IH && (unsigned)iw < (unsigned)IW) {
          const _Float16* p =
              X + ((((long)bimg[t] * IH + ih) * IW + iw) << cinLog2) + ci;
          v8h blo = *(const v8h*)p;
          v8h bhi = *(const v8h*)(p + 8);
#pragma unroll
          for (int i = 0; i < 8; ++i) { b[i] = blo[i]; b[8 + i] = bhi[i]; }
        }
      }
      acc[t] = __builtin_amdgcn_wmma_f32_16x16x32_f16(
          false, a, false, b, (short)0, acc[t], false, false);
    }
  }

  // ---- epilogue: BN + residual + ReLU, vectorized NHWC b128 stores -------
  float gm[8], bt[8];
#pragma unroll
  for (int rr = 0; rr < 8; ++rr) {
    const int m = mtile * 16 + 8 * half + rr;
    gm[rr] = gamma[m];
    bt[rr] = beta[m];
  }
#pragma unroll
  for (int t = 0; t < 4; ++t) {
    if (!nv[t]) continue;
    const long obase =
        ((((long)bimg[t] * OH + ohv[t]) * OW + owv[t]) * Cout) +
        mtile * 16 + 8 * half;                      // 8 contiguous channels
    v8h rsv{};
    if (Res) rsv = *(const v8h*)(Res + obase);
    v8h o;
#pragma unroll
    for (int rr = 0; rr < 8; ++rr) {
      float v = acc[t][rr] * gm[rr] + bt[rr];
      if (Res) v += (float)rsv[rr];
      if (relu) v = v > 0.f ? v : 0.f;
      o[rr] = (_Float16)v;
    }
    *(v8h*)(Y + obase) = o;
  }
}

// ---------------------------------------------------------------------------
// 3x3 s2 p1 maxpool, NHWC f16
// ---------------------------------------------------------------------------
__global__ void maxpool_kernel(const _Float16* __restrict__ X,
                               _Float16* __restrict__ Y,
                               int C, int IH, int IW, int OH, int OW) {
  int total = 8 * OH * OW * C;
  int i = blockIdx.x * blockDim.x + threadIdx.x;
  if (i >= total) return;
  int c  = i % C; int t = i / C;
  int ow = t % OW; t /= OW;
  int oh = t % OH;
  int b  = t / OH;
  float m = -3.4e38f;
  for (int r = 0; r < 3; ++r) {
    int ih = oh * 2 - 1 + r;
    if ((unsigned)ih >= (unsigned)IH) continue;
    for (int s = 0; s < 3; ++s) {
      int iw = ow * 2 - 1 + s;
      if ((unsigned)iw >= (unsigned)IW) continue;
      float v = (float)X[(((long)b * IH + ih) * IW + iw) * C + c];
      m = v > m ? v : m;
    }
  }
  Y[i] = (_Float16)m;
}

// ---------------------------------------------------------------------------
// fuse: features are NHWC [8,7,7,512]; out NCHW fp32 [8,1024,49,49]
// out[b,c,i,j] = c<512 ? fr[b, i/7, j/7, c] : fg[b, i%7, j%7, c-512]
// ---------------------------------------------------------------------------
__global__ void fuse_kernel(const _Float16* __restrict__ fr,
                            const _Float16* __restrict__ fg,
                            float* __restrict__ out) {
  const long total = 8L * 1024 * 49 * 49;
  long li = (long)blockIdx.x * blockDim.x + threadIdx.x;
  if (li >= total) return;
  int j  = (int)(li % 49); long t = li / 49;
  int ii = (int)(t % 49); t /= 49;
  int c  = (int)(t % 1024);
  int b  = (int)(t / 1024);
  float v;
  if (c < 512)
    v = (float)fr[((b * 7 + ii / 7) * 7 + j / 7) * 512 + c];
  else
    v = (float)fg[((b * 7 + ii % 7) * 7 + j % 7) * 512 + (c - 512)];
  out[li] = v;
}

// ---------------------------------------------------------------------------
// host orchestration
// ---------------------------------------------------------------------------
static inline unsigned gblocks(long n, int bs) { return (unsigned)((n + bs - 1) / bs); }

static void launch_pack(hipStream_t st, const float* W, _Float16* Wp,
                        int CinReal, int CinPad, int RS, int Kpad, long cout) {
  long total = cout * Kpad;
  pack_w_kernel<<<gblocks(total, 256), 256, 0, st>>>(W, Wp, CinReal, CinPad,
                                                     RS, Kpad, total);
}

static void launch_conv(hipStream_t st, const _Float16* X, const _Float16* Wp,
                        const float* g, const float* b, const _Float16* res,
                        _Float16* Y, int Cin, int IH, int IW, int Cout,
                        int OH, int OW, int R, int stride, int pad,
                        int Kprod, int Kpad, int relu) {
  int Ntot = 8 * OH * OW;
  long ntiles = (Ntot + 15) / 16;
  dim3 grid(gblocks(ntiles, 16), (unsigned)(Cout / 16), 1);
  int cl = 31 - __builtin_clz((unsigned)Cin);
  if (R == 1)
    conv_wmma_kernel<1, 1><<<grid, 128, 0, st>>>(X, Wp, g, b, res, Y, cl, IH,
                                                 IW, Cout, OH, OW, stride, pad,
                                                 Kprod, Kpad, relu);
  else if (R == 3)
    conv_wmma_kernel<3, 3><<<grid, 128, 0, st>>>(X, Wp, g, b, res, Y, cl, IH,
                                                 IW, Cout, OH, OW, stride, pad,
                                                 Kprod, Kpad, relu);
  else
    conv_wmma_kernel<7, 7><<<grid, 128, 0, st>>>(X, Wp, g, b, res, Y, cl, IH,
                                                 IW, Cout, OH, OW, stride, pad,
                                                 Kprod, Kpad, relu);
}

extern "C" void kernel_launch(void* const* d_in, const int* in_sizes, int n_in,
                              void* d_out, int out_size, void* d_ws, size_t ws_size,
                              hipStream_t stream) {
  (void)in_sizes; (void)out_size; (void)ws_size;
  const float* imgs[2] = {(const float*)d_in[0], (const float*)d_in[1]};

  // ---- ResNet-18 block configuration -------------------------------------
  static const int cs[8]      = {64, 64, 128, 128, 256, 256, 512, 512};
  static const int strides[8] = {1, 1, 2, 1, 2, 1, 2, 1};
  static const int hasds[8]   = {0, 0, 1, 0, 1, 0, 1, 0};
  static const int blk_base[8]= {0, 6, 12, 21, 27, 36, 42, 51};

  // ---- parameter leaf table (jax pytree order: sorted dict keys) ---------
  // top: "blocks" < "stem_b" < "stem_g" < "stem_w"
  // per block (no ds): b1,b2,g1,g2,w1,w2 ; (ds): b1,b2,bd,g1,g2,gd,w1,w2,wd
  long leaf_sz[60];
  {
    static const int ins[8] = {64, 64, 64, 128, 128, 256, 256, 512};
    int idx = 0;
    for (int bk = 0; bk < 8; ++bk) {
      long c = cs[bk], ic = ins[bk];
      if (!hasds[bk]) {
        leaf_sz[idx++] = c; leaf_sz[idx++] = c;
        leaf_sz[idx++] = c; leaf_sz[idx++] = c;
        leaf_sz[idx++] = c * ic * 9; leaf_sz[idx++] = c * c * 9;
      } else {
        leaf_sz[idx++] = c; leaf_sz[idx++] = c; leaf_sz[idx++] = c;
        leaf_sz[idx++] = c; leaf_sz[idx++] = c; leaf_sz[idx++] = c;
        leaf_sz[idx++] = c * ic * 9; leaf_sz[idx++] = c * c * 9;
        leaf_sz[idx++] = c * ic;
      }
    }
    leaf_sz[idx++] = 64;           // stem_b
    leaf_sz[idx++] = 64;           // stem_g
    leaf_sz[idx++] = 64L * 3 * 49; // stem_w
  }
  const float* leafp[2][60];
  if (n_in >= 2 + 120) {
    for (int net = 0; net < 2; ++net)
      for (int i = 0; i < 60; ++i)
        leafp[net][i] = (const float*)d_in[2 + net * 60 + i];
  } else {
    for (int net = 0; net < 2; ++net) {
      const float* base = (const float*)d_in[2 + net];
      long off = 0;
      for (int i = 0; i < 60; ++i) { leafp[net][i] = base + off; off += leaf_sz[i]; }
    }
  }

  // ---- workspace carve ----------------------------------------------------
  const long ACT_ELEMS = 8L * 224 * 224 * 16;  // 6,422,528 (== 8*112*112*64)
  const long WP_ELEMS  = 4608L * 512;          // largest packed weight
  const long FT_ELEMS  = 8L * 7 * 7 * 512;     // final features (NHWC)
  char* ws = (char*)d_ws;
  auto take = [&](long bytes) { char* p = ws; ws += (bytes + 255) & ~255L; return p; };
  _Float16* ACT0 = (_Float16*)take(ACT_ELEMS * 2);
  _Float16* ACT1 = (_Float16*)take(ACT_ELEMS * 2);
  _Float16* ACT2 = (_Float16*)take(ACT_ELEMS * 2);
  _Float16* WP   = (_Float16*)take(WP_ELEMS * 2);
  _Float16* FEAT[2];
  FEAT[0] = (_Float16*)take(FT_ELEMS * 2);
  FEAT[1] = (_Float16*)take(FT_ELEMS * 2);

  // ---- run both networks --------------------------------------------------
  for (int net = 0; net < 2; ++net) {
    const float* const* L = leafp[net];

    // input NCHW fp32 -> NHWC f16 with Cin padded 3->16
    cvt_input_kernel<<<gblocks(8L * 224 * 224, 256), 256, 0, stream>>>(
        imgs[net], ACT2);

    // stem: 7x7 s2 p3, (16)->64, BN+ReLU   (K = 49*16 = 784, Kpad = 800)
    launch_pack(stream, L[59], WP, 3, 16, 49, 800, 64);
    launch_conv(stream, ACT2, WP, L[58], L[57], nullptr, ACT0,
                16, 224, 224, 64, 112, 112, 7, 2, 3, 784, 800, 1);

    // maxpool 3x3 s2 p1: [8,112,112,64] -> [8,56,56,64]
    maxpool_kernel<<<gblocks(8L * 56 * 56 * 64, 256), 256, 0, stream>>>(
        ACT0, ACT1, 64, 112, 112, 56, 56);

    _Float16 *bx = ACT1, *ba = ACT0, *bb = ACT2;  // x, temp, idn/out
    int inC = 64, inHW = 56;
    for (int bk = 0; bk < 8; ++bk) {
      const int c = cs[bk], st = strides[bk];
      const int oHW = (inHW + 2 - 3) / st + 1;
      const int base = blk_base[bk];
      if (hasds[bk]) {
        // idn = BN(conv1x1 s2 (x)) -> bb          (K = inC)
        launch_pack(stream, L[base + 8], WP, inC, inC, 1, inC, c);
        launch_conv(stream, bx, WP, L[base + 5], L[base + 2], nullptr, bb,
                    inC, inHW, inHW, c, oHW, oHW, 1, st, 0, inC, inC, 0);
        // y = ReLU(BN(conv3x3 s (x))) -> ba       (K = 9*inC)
        launch_pack(stream, L[base + 6], WP, inC, inC, 9, inC * 9, c);
        launch_conv(stream, bx, WP, L[base + 3], L[base + 0], nullptr, ba,
                    inC, inHW, inHW, c, oHW, oHW, 3, st, 1, inC * 9, inC * 9, 1);
        // x = ReLU(BN(conv3x3 (y)) + idn) -> bx   (K = 9*c)
        launch_pack(stream, L[base + 7], WP, c, c, 9, c * 9, c);
        launch_conv(stream, ba, WP, L[base + 4], L[base + 1], bb, bx,
                    c, oHW, oHW, c, oHW, oHW, 3, 1, 1, c * 9, c * 9, 1);
      } else {
        // y = ReLU(BN(conv3x3 (x))) -> ba
        launch_pack(stream, L[base + 4], WP, inC, inC, 9, inC * 9, c);
        launch_conv(stream, bx, WP, L[base + 2], L[base + 0], nullptr, ba,
                    inC, inHW, inHW, c, oHW, oHW, 3, st, 1, inC * 9, inC * 9, 1);
        // x = ReLU(BN(conv3x3 (y)) + x) -> bb ; rotate
        launch_pack(stream, L[base + 5], WP, c, c, 9, c * 9, c);
        launch_conv(stream, ba, WP, L[base + 3], L[base + 1], bx, bb,
                    c, oHW, oHW, c, oHW, oHW, 3, 1, 1, c * 9, c * 9, 1);
        _Float16* tmp = bx; bx = bb; bb = tmp;
      }
      inC = c; inHW = oHW;
    }
    // stash NHWC [8,7,7,512] features
    (void)hipMemcpyAsync(FEAT[net], bx, FT_ELEMS * sizeof(_Float16),
                         hipMemcpyDeviceToDevice, stream);
  }

  // ---- fuse -> d_out fp32 [8,1024,49,49] ---------------------------------
  fuse_kernel<<<gblocks(8L * 1024 * 49 * 49, 256), 256, 0, stream>>>(
      FEAT[0], FEAT[1], (float*)d_out);
}
